// SpikingCNN_5514738008452
// MI455X (gfx1250) — matrix operand order repaired and checked by reading, not compiled, not run on hardware
//
#include <hip/hip_runtime.h>
#include <stdint.h>

// ---------- CDNA5 vector types ----------
typedef _Float16 v16h __attribute__((ext_vector_type(16)));
typedef float    v8f  __attribute__((ext_vector_type(8)));
typedef unsigned int u32x4 __attribute__((ext_vector_type(4)));
typedef int          i32x8 __attribute__((ext_vector_type(8)));
typedef int          i32x4 __attribute__((ext_vector_type(4)));

union HFrag { uint32_t u[8]; v16h h; };

// =====================================================================
// Conv1d (im2col GEMM via v_wmma_f32_16x16x32_f16) + bias + BN-stat
// accumulation.  One workgroup = one batch element x 64 output t's.
// Weights loaded into LDS by the Tensor Data Mover (TDM).
// Output layout: (Tconv, B, COUT) so the LIF scan is coalesced.
// =====================================================================
template<int CIN, int KSZ, int STRIDE, int PAD, int COUT, bool IN_BCT>
__global__ __launch_bounds__(128)
void conv_bn_stats(const float* __restrict__ in, const float* __restrict__ W,
                   const float* __restrict__ bias, float* __restrict__ out,
                   float* __restrict__ stats, int B, int Tin)
{
    constexpr int KDIM  = CIN * KSZ;               // GEMM K (im2col)
    constexpr int KDIMP = (KDIM + 31) & ~31;       // padded to WMMA K=32
    constexpr int MT    = COUT / 16;               // M tiles
    constexpr int WN    = COUT * KDIM;             // raw weight element count

    __shared__ float    wstage[WN];                // TDM destination (raw f32)
    __shared__ _Float16 Wl[COUT * KDIMP];          // f16 weight panel [co][k]
    __shared__ _Float16 Bl[64 * KDIMP];            // f16 im2col panel [n][k]
    __shared__ float    ssum[COUT], ssq[COUT];     // per-channel BN partials

    const int tid = threadIdx.x;
    const int b   = blockIdx.y;
    const int t0  = blockIdx.x * 64;

    if (tid < COUT) { ssum[tid] = 0.f; ssq[tid] = 0.f; }

    // ---- TDM: DMA the raw f32 weights global->LDS (1-D tile) ----
    if (tid == 0) {
        const uint64_t ga = (uint64_t)(uintptr_t)W;
        const uint32_t la = (uint32_t)(uintptr_t)wstage;     // LDS byte addr
        u32x4 g0;
        g0[0] = 1u;                                          // count=1 (valid D#)
        g0[1] = la;                                          // lds_addr
        g0[2] = (uint32_t)ga;                                // global_addr[31:0]
        g0[3] = ((uint32_t)(ga >> 32) & 0x01FFFFFFu)         // global_addr[56:32]
              | 0x80000000u;                                 // type=2 ("image")
        i32x8 g1;
        g1[0] = (int)(2u << 16);                             // data_size=4B
        g1[1] = (int)(((uint32_t)WN & 0xFFFFu) << 16);       // tensor_dim0 lo
        g1[2] = (int)(((uint32_t)WN >> 16) | (1u << 16));    // dim0 hi | tensor_dim1=1
        g1[3] = (int)(((uint32_t)WN & 0xFFFFu) << 16);       // tile_dim0 = WN
        g1[4] = 1;                                           // tile_dim1=1
        g1[5] = WN;                                          // tensor_dim0_stride
        g1[6] = 0; g1[7] = 0;
        i32x4 z4 = {0, 0, 0, 0};
        i32x8 z8 = {0, 0, 0, 0, 0, 0, 0, 0};
        __builtin_amdgcn_tensor_load_to_lds(g0, g1, z4, z4, z8, 0);
        __builtin_amdgcn_s_wait_tensorcnt(0);
    }
    __syncthreads();

    // ---- f32 -> f16, zero-padded weight panel [co][KDIMP] ----
    for (int idx = tid; idx < COUT * KDIMP; idx += 128) {
        const int co = idx / KDIMP, kk = idx % KDIMP;
        const float v = (kk < KDIM) ? wstage[co * KDIM + kk] : 0.f;
        Wl[idx] = (_Float16)v;
    }
    // ---- im2col panel [n][KDIMP], n = output t within this WG ----
    for (int idx = tid; idx < 64 * KDIMP; idx += 128) {
        const int n = idx / KDIMP, kk = idx % KDIMP;
        float v = 0.f;
        if (kk < KDIM) {
            const int ci = kk / KSZ, k = kk % KSZ;
            const int ti = STRIDE * (t0 + n) + k - PAD;
            if (ti >= 0 && ti < Tin)
                v = IN_BCT ? in[((size_t)b * CIN + ci) * (size_t)Tin + ti]
                           : in[((size_t)ti * B + b) * CIN + ci];
        }
        Bl[idx] = (_Float16)v;
    }
    __syncthreads();

    const int lane  = tid & 31;
    const int wave  = tid >> 5;
    const int mhalf = lane >> 4;     // lanes 16-31 hold the other K/M half
    const int lan15 = lane & 15;

    // 4 waves cover MT x 4 tiles of 16(co) x 16(t)
    for (int tile = wave; tile < MT * 4; tile += 4) {
        const int mt = tile % MT, nt = tile / MT;
        const int coBase = mt * 16, nBase = nt * 16;
        v8f acc = {};
        for (int kb = 0; kb < KDIMP; kb += 32) {
            HFrag a, bf;
#pragma unroll
            for (int v = 0; v < 8; ++v) {
                // A 16x32 f16 layout: lanes 0-15/16-31 -> same M rows, K halves
                const int ka  = (v < 4) ? (mhalf * 8 + 2 * v)
                                        : (16 + mhalf * 8 + 2 * (v - 4));
                a.u[v]  = *(const uint32_t*)&Wl[(coBase + lan15) * KDIMP + kb + ka];
                // B 32x16 f16 layout: N = lane&15, K order same as A
                const int kbv = (v < 4) ? (mhalf * 8 + 2 * v)
                                        : (16 + mhalf * 8 + 2 * (v - 4));
                bf.u[v] = *(const uint32_t*)&Bl[(nBase + lan15) * KDIMP + kb + kbv];
            }
            acc = __builtin_amdgcn_wmma_f32_16x16x32_f16(
                      false, a.h, false, bf.h, (short)0, acc, false, false);
        }
        // D layout: VGPR r -> co = coBase + r + 8*mhalf, N = lane&15
        const int t = t0 + nBase + lan15;
#pragma unroll
        for (int r = 0; r < 8; ++r) {
            const int co = coBase + r + 8 * mhalf;
            const float val = acc[r] + bias[co];
            out[((size_t)t * B + b) * COUT + co] = val;
            atomicAdd(&ssum[co], val);          // ds_add_f32
            atomicAdd(&ssq[co],  val * val);
        }
    }
    __syncthreads();
    if (tid < COUT) {
        atomicAdd(&stats[tid],        ssum[tid]);   // global_atomic_add_f32
        atomicAdd(&stats[COUT + tid], ssq[tid]);
    }
}

// =====================================================================
// BN finalize: scale = g*rsqrt(var+eps), shift = beta - mean*scale
// =====================================================================
__global__ void bn_finalize(const float* __restrict__ stats,
                            const float* __restrict__ gamma,
                            const float* __restrict__ beta,
                            float* __restrict__ scsh, int C, float invN)
{
    const int c = threadIdx.x;
    if (c < C) {
        const float mean = stats[c] * invN;
        const float var  = stats[C + c] * invN - mean * mean;
        const float sc   = gamma[c] * rsqrtf(var + 1e-5f);
        scsh[c]     = sc;
        scsh[C + c] = beta[c] - mean * sc;
    }
}

// =====================================================================
// LIF scan (beta=0.9, subtract reset) + maxpool2.  Time split into
// 2048-step chunks with 128-step warm-up replay (0.9^128 ~ 2e-6), so
// every stage exposes 4096 independent scan lanes.
// cur layout (Tconv,B,C); output (T/2,B,C) or final (B,C,T/2).
// =====================================================================
#define LIF_CHUNK 2048
#define LIF_WARM  128

template<bool OUT_BCT>
__global__ __launch_bounds__(256)
void lif_pool(const float* __restrict__ cur, const float* __restrict__ scsh,
              float* __restrict__ out, int B, int C, int Tconv)
{
    const int gid = blockIdx.x * blockDim.x + threadIdx.x;
    const int bc  = B * C;
    const int chunkIdx = gid / bc;
    const int rem = gid - chunkIdx * bc;
    const int b = rem / C, c = rem % C;
    if (chunkIdx >= Tconv / LIF_CHUNK) return;

    const float scale = scsh[c], shift = scsh[C + c];
    const size_t strideT = (size_t)B * C;
    const float* base = cur + (size_t)rem;

    float mem = 0.f;
    const int tstart = chunkIdx * LIF_CHUNK;
    int w0 = tstart - LIF_WARM; if (w0 < 0) w0 = 0;
    for (int t = w0; t < tstart; ++t) {                 // warm-up (no output)
        const float inp = base[(size_t)t * strideT] * scale + shift;
        const float rst = (mem > 1.0f) ? 1.0f : 0.0f;
        mem = 0.9f * mem + inp - rst;
    }
    const int Tp = Tconv >> 1;
    for (int t = tstart; t < tstart + LIF_CHUNK; t += 2) {
        const float i0 = base[(size_t)t * strideT] * scale + shift;
        const float r0 = (mem > 1.0f) ? 1.0f : 0.0f;
        mem = 0.9f * mem + i0 - r0;
        const float s0 = (mem > 1.0f) ? 1.0f : 0.0f;
        const float i1 = base[(size_t)(t + 1) * strideT] * scale + shift;
        const float r1 = (mem > 1.0f) ? 1.0f : 0.0f;
        mem = 0.9f * mem + i1 - r1;
        const float s1 = (mem > 1.0f) ? 1.0f : 0.0f;
        const float p  = fmaxf(s0, s1);
        const int tp = t >> 1;
        if (OUT_BCT) out[((size_t)b * C + c) * Tp + tp] = p;
        else         out[((size_t)tp * B + b) * C + c]  = p;
    }
}

__global__ void zero_ws(float* __restrict__ p, int n)
{
    const int i = blockIdx.x * blockDim.x + threadIdx.x;
    if (i < n) p[i] = 0.f;
}

// =====================================================================
extern "C" void kernel_launch(void* const* d_in, const int* in_sizes, int n_in,
                              void* d_out, int out_size, void* d_ws, size_t ws_size,
                              hipStream_t stream)
{
    (void)in_sizes; (void)n_in; (void)out_size; (void)ws_size;
    const float* x   = (const float*)d_in[0];
    const float* w1  = (const float*)d_in[1];  const float* b1  = (const float*)d_in[2];
    const float* g1  = (const float*)d_in[3];  const float* be1 = (const float*)d_in[4];
    const float* w2  = (const float*)d_in[5];  const float* b2  = (const float*)d_in[6];
    const float* g2  = (const float*)d_in[7];  const float* be2 = (const float*)d_in[8];
    const float* w3  = (const float*)d_in[9];  const float* b3  = (const float*)d_in[10];
    const float* g3  = (const float*)d_in[11]; const float* be3 = (const float*)d_in[12];

    float* ws     = (float*)d_ws;
    float* stats1 = ws;        float* stats2 = ws + 128;  float* stats3 = ws + 256;
    float* scsh1  = ws + 384;  float* scsh2  = ws + 512;  float* scsh3  = ws + 640;
    float* conv1o = ws + 768;                                   // (16384,32,16)
    float* pool1  = conv1o + (size_t)16384 * 32 * 16;           // ( 8192,32,16)
    float* conv2o = pool1  + (size_t)8192  * 32 * 16;           // ( 8192,32,32)
    float* pool2  = conv2o + (size_t)8192  * 32 * 32;           // ( 4096,32,32)
    float* conv3o = pool2  + (size_t)4096  * 32 * 32;           // ( 4096,32,64)

    const int B = 32;
    zero_ws<<<3, 256, 0, stream>>>(ws, 768);

    // Block 1: conv(2->16,K5,s4,p2) -> BN -> LIF -> pool
    conv_bn_stats<2, 5, 4, 2, 16, true>
        <<<dim3(16384 / 64, B), 128, 0, stream>>>(x, w1, b1, conv1o, stats1, B, 65536);
    bn_finalize<<<1, 64, 0, stream>>>(stats1, g1, be1, scsh1, 16, 1.0f / (B * 16384.0f));
    lif_pool<false><<<16, 256, 0, stream>>>(conv1o, scsh1, pool1, B, 16, 16384);

    // Block 2: conv(16->32,K5,s1,p2)
    conv_bn_stats<16, 5, 1, 2, 32, false>
        <<<dim3(8192 / 64, B), 128, 0, stream>>>(pool1, w2, b2, conv2o, stats2, B, 8192);
    bn_finalize<<<1, 64, 0, stream>>>(stats2, g2, be2, scsh2, 32, 1.0f / (B * 8192.0f));
    lif_pool<false><<<16, 256, 0, stream>>>(conv2o, scsh2, pool2, B, 32, 8192);

    // Block 3: conv(32->64,K3,s1,p1) -> final output (B,64,2048)
    conv_bn_stats<32, 3, 1, 1, 64, false>
        <<<dim3(4096 / 64, B), 128, 0, stream>>>(pool2, w3, b3, conv3o, stats3, B, 4096);
    bn_finalize<<<1, 64, 0, stream>>>(stats3, g3, be3, scsh3, 64, 1.0f / (B * 4096.0f));
    lif_pool<true><<<16, 256, 0, stream>>>(conv3o, scsh3, (float*)d_out, B, 64, 4096);
}
